// RSSM_19842748908134
// MI455X (gfx1250) — matrix operand order repaired and checked by reading, not compiled, not run on hardware
//
#include <hip/hip_runtime.h>

// ---------------------------------------------------------------------------
// RSSM observe loop for MI455X (gfx1250).
//  * All weights fp32->bf16 ONCE (178MB -> resident in 192MB L2), packed in
//    WMMA B-fragment order.
//  * Activations packed to bf16 A-fragment order once per matrix per step,
//    so the GEMM hot loop is: load A frag (32B) + load B frag (32B) + v_wmma.
//  * M=16 (batch) == WMMA tile M; one wave per 16x16 output tile, full-K acc.
// ---------------------------------------------------------------------------

#define BB 16
#define LL 32
#define DETER 8192
#define HIDDEN 1024
#define FLATD 2048      // STOCH*DISC
#define BLOCKS 8
#define ACT_DIM 32
#define EMBED 1024
#define FEATW 12288     // 2048 + 8192 + 2048

typedef __attribute__((ext_vector_type(16))) __bf16 v16bf;
typedef __attribute__((ext_vector_type(8)))  float  v8f;

static __device__ __forceinline__ unsigned short f2bf(float f) {
    unsigned int u = __builtin_bit_cast(unsigned int, f);
    u += 0x7FFFu + ((u >> 16) & 1u);          // round-to-nearest-even
    return (unsigned short)(u >> 16);
}
static __device__ __forceinline__ float sigm(float x) { return 1.0f / (1.0f + __expf(-x)); }

// ---------------------------------------------------------------------------
// Pack fp32 row-major W[blk][K][N] -> bf16 WMMA B-fragments.
// Fragment order: [blk][nt*ktiles + kt][lane][16], ISA dense 16-bit B 32x16:
// lane -> N = lane&15, elem e -> K = 8*(lane>>4) + (e&7) + 16*(e>>3).
// ---------------------------------------------------------------------------
__global__ __launch_bounds__(32)
void pack_w(const float* __restrict__ W, unsigned short* __restrict__ Wp, int K, int N) {
    const int frag  = blockIdx.x;
    const int blk   = blockIdx.y;
    const int ktiles = K >> 5;
    const int nt = frag / ktiles, kt = frag % ktiles;
    const int lane = threadIdx.x;
    const float* Wb = W + (long long)blk * K * N;
    unsigned short* out = Wp + (((long long)blk * gridDim.x + frag) * 32 + lane) * 16;
    const int n = nt * 16 + (lane & 15);
    const int kb = kt * 32 + 8 * (lane >> 4);
#pragma unroll
    for (int e = 0; e < 16; ++e)
        out[e] = f2bf(Wb[(long long)(kb + (e & 7) + 16 * (e >> 3)) * N + n]);
}

// ---------------------------------------------------------------------------
// Pack fp32 row-major A[blk][16][K] (row stride lda, block offset aBlk) into
// bf16 WMMA A-fragment order: [blk][kt][lane][16], ISA 16-bit A 16x32:
// lane -> M = lane&15; elems 0..7 -> K = 8*hi + e; elems 8..15 -> K = 8*hi+16+(e-8).
// One wave per k-tile; runs ONCE per activation matrix (vs once per n-tile).
// ---------------------------------------------------------------------------
__global__ __launch_bounds__(32)
void pack_a(const float* __restrict__ A, long long aBlk, int lda,
            unsigned short* __restrict__ Ap) {
    const int kt = blockIdx.x, blk = blockIdx.y, lane = threadIdx.x;
    const int hi = lane >> 4, m = lane & 15;
    const float* p0 = A + (long long)blk * aBlk + (long long)m * lda + kt * 32 + 8 * hi;
    unsigned short* out = Ap + (long long)blk * gridDim.x * 512
                        + ((long long)kt * 32 + lane) * 16;
    float tf[16];
    *(float4*)(tf + 0)  = *(const float4*)(p0 + 0);
    *(float4*)(tf + 4)  = *(const float4*)(p0 + 4);
    *(float4*)(tf + 8)  = *(const float4*)(p0 + 16);
    *(float4*)(tf + 12) = *(const float4*)(p0 + 20);
#pragma unroll
    for (int e = 0; e < 16; ++e) out[e] = f2bf(tf[e]);
}

// ---------------------------------------------------------------------------
// Y[16 x Nblk] (per block) = A @ W + bias, A/W pre-packed bf16 fragments.
// Hot loop: 2x 32B loads + v_wmma_f32_16x16x32_bf16 (+ prefetch hint).
// ---------------------------------------------------------------------------
__global__ __launch_bounds__(32)
void wmma_gemm(const unsigned short* __restrict__ Ap, long long aBlk,
               const unsigned short* __restrict__ Wp, long long wBlk,
               const float* __restrict__ bias,
               float* __restrict__ Y, long long yBlk, int ldy, int K) {
    const int lane = threadIdx.x;
    const int nt   = blockIdx.x;
    const int blk  = blockIdx.y;
    const int Nblk = gridDim.x * 16;
    const int hi   = lane >> 4;
    const int ktiles = K >> 5;

    const v16bf* Af = (const v16bf*)(Ap + (long long)blk * aBlk) + lane;
    const v16bf* Wf = (const v16bf*)(Wp + (long long)blk * wBlk
                                        + (long long)nt * ktiles * 512) + lane;
    v8f c = {};
    for (int kt = 0; kt < ktiles; ++kt) {
        __builtin_prefetch(Wf + (long long)(kt + 1) * 32, 0, 1);  // global_prefetch_b8
        v16bf a = Af[(long long)kt * 32];
        v16bf b = Wf[(long long)kt * 32];
        c = __builtin_amdgcn_wmma_f32_16x16x32_bf16(false, a, false, b,
                                                    (short)0, c, false, false);
    }
    const int col = nt * 16 + (lane & 15);
    const float bval = bias ? bias[blk * Nblk + col] : 0.0f;
    float* Yp = Y + (long long)blk * yBlk + col;
#pragma unroll
    for (int r = 0; r < 8; ++r)                 // C/D layout: vgpr r -> M = r + 8*hi
        Yp[(long long)(r + 8 * hi) * ldy] = c[r] + bval;
}

// ---------------------------------------------------------------------------
// In-place RMS-norm (+eps) * g, then SiLU, over `width` cols of one row.
// ---------------------------------------------------------------------------
__global__ __launch_bounds__(256)
void rms_silu(float* __restrict__ X, int ldx, const float* __restrict__ g, int width) {
    __shared__ float red[8];
    float* x = X + (long long)blockIdx.x * ldx;
    float ss = 0.0f;
    for (int i = threadIdx.x; i < width; i += 256) { float v = x[i]; ss += v * v; }
    for (int o = 16; o > 0; o >>= 1) ss += __shfl_down(ss, o, 32);
    if ((threadIdx.x & 31) == 0) red[threadIdx.x >> 5] = ss;
    __syncthreads();
    if (threadIdx.x == 0) {
        float t = 0.0f;
        for (int w = 0; w < 8; ++w) t += red[w];
        red[0] = rsqrtf(t / (float)width + 1e-4f);
    }
    __syncthreads();
    const float s = red[0];
    for (int i = threadIdx.x; i < width; i += 256) {
        float v = x[i] * s * g[i];
        x[i] = v * sigm(v);
    }
}

// Reset-mask deter/stoch, normalize action.  grid = 16*8192/256.
__global__ __launch_bounds__(256)
void prep(float* __restrict__ deter, float* __restrict__ stoch, float* __restrict__ actn,
          const float* __restrict__ action, const int* __restrict__ reset, int t) {
    int i = blockIdx.x * 256 + threadIdx.x;
    int b = i >> 13, j = i & 8191;
    bool rst = reset[b * LL + t] != 0;
    if (rst) deter[i] = 0.0f;
    if (j < FLATD && rst) stoch[b * FLATD + j] = 0.0f;
    if (j < ACT_DIM) {
        float a = rst ? 0.0f : action[((long long)b * LL + t) * ACT_DIM + j];
        actn[b * ACT_DIM + j] = a / fmaxf(fabsf(a), 1.0f);
    }
}

// xcat[k][b][0:1024]=deter block k, [1024:4096]=x.  grid = 8*16*4096/256.
__global__ __launch_bounds__(256)
void build_xcat(float* __restrict__ xcat, const float* __restrict__ deter,
                const float* __restrict__ xbuf) {
    int i = blockIdx.x * 256 + threadIdx.x;
    int k = i / (BB * 4096), r = i % (BB * 4096);
    int b = r / 4096, c = r % 4096;
    xcat[i] = (c < 1024) ? deter[b * DETER + k * 1024 + c]
                         : xbuf[b * 3072 + (c - 1024)];
}

// GRU gates -> new deter; also feed catbuf[:, :8192] and feat deter slice.
__global__ __launch_bounds__(256)
void gru_update(const float* __restrict__ gbuf, float* __restrict__ deter,
                float* __restrict__ catbuf, float* __restrict__ feat, int t) {
    int i = blockIdx.x * 256 + threadIdx.x;
    int b = i >> 13, j = i & 8191;
    int blk = j >> 10, o = j & 1023;
    const float* g = gbuf + (long long)b * 24576 + blk * 3072;
    float r  = sigm(g[o]);
    float cc = g[1024 + o];
    float u  = sigm(g[2048 + o] - 1.0f);
    float cl = tanhf(r * cc);
    float d  = u * cl + (1.0f - u) * deter[i];
    deter[i] = d;
    catbuf[(long long)b * 9216 + j] = d;
    feat[((long long)b * LL + t) * FEATW + FLATD + j] = d;
}

__global__ __launch_bounds__(256)
void copy_embed(float* __restrict__ catbuf, const float* __restrict__ embed, int t) {
    int i = blockIdx.x * 256 + threadIdx.x;     // 16*1024
    int b = i >> 10, j = i & 1023;
    catbuf[(long long)b * 9216 + DETER + j] = embed[((long long)b * LL + t) * EMBED + j];
}

// Softmax + unimix + Gumbel-argmax sample (deterministic hash RNG).
__global__ __launch_bounds__(64)
void sample_k(const float* __restrict__ logitbuf, float* __restrict__ stoch,
              float* __restrict__ feat, int t) {
    __shared__ float sv[64];
    __shared__ int   si[64];
    int b = blockIdx.x, s = blockIdx.y, d = threadIdx.x;
    float lg = logitbuf[(long long)b * FLATD + s * 64 + d];
    sv[d] = lg; __syncthreads();
    for (int o = 32; o > 0; o >>= 1) { if (d < o) sv[d] = fmaxf(sv[d], sv[d + o]); __syncthreads(); }
    float mx = sv[0]; __syncthreads();
    float e = __expf(lg - mx);
    sv[d] = e; __syncthreads();
    for (int o = 32; o > 0; o >>= 1) { if (d < o) sv[d] += sv[d + o]; __syncthreads(); }
    float p = 0.01f / 64.0f + 0.99f * (e / sv[0]);
    __syncthreads();
    unsigned h = (unsigned)((((t * BB + b) * 32 + s) * 64) + d) * 0x9E3779B9u + 0x85EBCA6Bu;
    h ^= h >> 16; h *= 0x7FEB352Du; h ^= h >> 15; h *= 0x846CA68Bu; h ^= h >> 16;
    float u01 = (float)((h >> 8) + 1u) * (1.0f / 16777216.0f);
    float gv = __logf(p) - __logf(-__logf(u01));
    sv[d] = gv; si[d] = d; __syncthreads();
    for (int o = 32; o > 0; o >>= 1) {
        if (d < o && sv[d + o] > sv[d]) { sv[d] = sv[d + o]; si[d] = si[d + o]; }
        __syncthreads();
    }
    int amax = si[0];
    long long fb = ((long long)b * LL + t) * FEATW;
    feat[fb + FLATD + DETER + s * 64 + d] = lg;          // logit slice
    float oh = (d == amax) ? 1.0f : 0.0f;                // straight-through fwd = one-hot
    feat[fb + s * 64 + d] = oh;                          // stoch slice
    stoch[b * FLATD + s * 64 + d] = oh;                  // carried state
}

// ---------------------------------------------------------------------------
extern "C" void kernel_launch(void* const* d_in, const int* in_sizes, int n_in,
                              void* d_out, int out_size, void* d_ws, size_t ws_size,
                              hipStream_t stream) {
    (void)in_sizes; (void)n_in; (void)out_size; (void)ws_size;
    const float* embed   = (const float*)d_in[0];
    const float* action  = (const float*)d_in[1];
    const int*   reset   = (const int*)  d_in[2];
    const float* init_st = (const float*)d_in[3];
    const float* init_de = (const float*)d_in[4];
    const float* in0_w = (const float*)d_in[5];
    const float* in0_b = (const float*)d_in[6];
    const float* in0_g = (const float*)d_in[7];
    const float* in1_w = (const float*)d_in[8];
    const float* in1_b = (const float*)d_in[9];
    const float* in1_g = (const float*)d_in[10];
    const float* in2_w = (const float*)d_in[11];
    const float* in2_b = (const float*)d_in[12];
    const float* in2_g = (const float*)d_in[13];
    const float* hid0_w = (const float*)d_in[14];
    const float* hid0_b = (const float*)d_in[15];
    const float* hid0_g = (const float*)d_in[16];
    const float* hid1_w = (const float*)d_in[17];
    const float* hid1_b = (const float*)d_in[18];
    const float* hid1_g = (const float*)d_in[19];
    const float* gru_w  = (const float*)d_in[20];
    const float* gru_b  = (const float*)d_in[21];
    const float* obs0_w = (const float*)d_in[22];
    const float* obs0_b = (const float*)d_in[23];
    const float* obs0_g = (const float*)d_in[24];
    const float* obsL_w = (const float*)d_in[25];
    const float* obsL_b = (const float*)d_in[26];
    float* feat = (float*)d_out;

    char* base = (char*)d_ws;
    size_t off = 0;
    auto take = [&](size_t bytes) -> char* {
        char* p = base + off;
        off = (off + bytes + 255) & ~(size_t)255;
        return p;
    };
    // packed bf16 weights (~178 MB total -> resident in the 192 MB L2)
    unsigned short* in0p  = (unsigned short*)take((size_t)DETER * HIDDEN * 2);
    unsigned short* in1p  = (unsigned short*)take((size_t)FLATD * HIDDEN * 2);
    unsigned short* in2p  = (unsigned short*)take((size_t)ACT_DIM * HIDDEN * 2);
    unsigned short* hid0p = (unsigned short*)take((size_t)BLOCKS * 4096 * 1024 * 2);
    unsigned short* hid1p = (unsigned short*)take((size_t)BLOCKS * 1024 * 1024 * 2);
    unsigned short* grup  = (unsigned short*)take((size_t)BLOCKS * 1024 * 3072 * 2);
    unsigned short* obs0p = (unsigned short*)take((size_t)9216 * 1024 * 2);
    unsigned short* obsLp = (unsigned short*)take((size_t)1024 * 2048 * 2);
    // fp32 activations / state
    float* deter  = (float*)take((size_t)BB * DETER * 4);
    float* stoch  = (float*)take((size_t)BB * FLATD * 4);
    float* actn   = (float*)take((size_t)BB * ACT_DIM * 4);
    float* xbuf   = (float*)take((size_t)BB * 3072 * 4);
    float* xcat   = (float*)take((size_t)BLOCKS * BB * 4096 * 4);
    float* hbuf   = (float*)take((size_t)BB * DETER * 4);
    float* h2buf  = (float*)take((size_t)BB * DETER * 4);
    float* gbuf   = (float*)take((size_t)BB * 24576 * 4);
    float* catbuf = (float*)take((size_t)BB * 9216 * 4);
    float* hob    = (float*)take((size_t)BB * 1024 * 4);
    float* lbuf   = (float*)take((size_t)BB * FLATD * 4);
    // bf16 packed activation fragments
    unsigned short* deterA = (unsigned short*)take((size_t)BB * DETER * 2);
    unsigned short* stochA = (unsigned short*)take((size_t)BB * FLATD * 2);
    unsigned short* actA   = (unsigned short*)take((size_t)BB * ACT_DIM * 2);
    unsigned short* xcatA  = (unsigned short*)take((size_t)BLOCKS * BB * 4096 * 2);
    unsigned short* hA     = (unsigned short*)take((size_t)BLOCKS * BB * 1024 * 2);
    unsigned short* h2A    = (unsigned short*)take((size_t)BLOCKS * BB * 1024 * 2);
    unsigned short* catA   = (unsigned short*)take((size_t)BB * 9216 * 2);
    unsigned short* hobA   = (unsigned short*)take((size_t)BB * 1024 * 2);

    hipMemcpyAsync(deter, init_de, (size_t)BB * DETER * 4, hipMemcpyDeviceToDevice, stream);
    hipMemcpyAsync(stoch, init_st, (size_t)BB * FLATD * 4, hipMemcpyDeviceToDevice, stream);

    // one-time fp32 -> packed bf16 weight conversion
    pack_w<<<dim3(64 * 256, 1), 32, 0, stream>>>(in0_w,  in0p,  DETER, 1024);
    pack_w<<<dim3(64 * 64,  1), 32, 0, stream>>>(in1_w,  in1p,  FLATD, 1024);
    pack_w<<<dim3(64 * 1,   1), 32, 0, stream>>>(in2_w,  in2p,  32,    1024);
    pack_w<<<dim3(64 * 128, 8), 32, 0, stream>>>(hid0_w, hid0p, 4096,  1024);
    pack_w<<<dim3(64 * 32,  8), 32, 0, stream>>>(hid1_w, hid1p, 1024,  1024);
    pack_w<<<dim3(192 * 32, 8), 32, 0, stream>>>(gru_w,  grup,  1024,  3072);
    pack_w<<<dim3(64 * 288, 1), 32, 0, stream>>>(obs0_w, obs0p, 9216,  1024);
    pack_w<<<dim3(128 * 32, 1), 32, 0, stream>>>(obsL_w, obsLp, 1024,  2048);

    for (int t = 0; t < LL; ++t) {
        prep<<<512, 256, 0, stream>>>(deter, stoch, actn, action, reset, t);
        pack_a<<<dim3(256, 1), 32, 0, stream>>>(deter, 0, DETER, deterA);
        pack_a<<<dim3(64,  1), 32, 0, stream>>>(stoch, 0, FLATD, stochA);
        pack_a<<<dim3(1,   1), 32, 0, stream>>>(actn,  0, ACT_DIM, actA);
        // input GEMMs -> xbuf[16 x 3072]
        wmma_gemm<<<dim3(64, 1), 32, 0, stream>>>(deterA, 0, in0p, 0, in0_b, xbuf + 0,    0, 3072, DETER);
        wmma_gemm<<<dim3(64, 1), 32, 0, stream>>>(stochA, 0, in1p, 0, in1_b, xbuf + 1024, 0, 3072, FLATD);
        wmma_gemm<<<dim3(64, 1), 32, 0, stream>>>(actA,   0, in2p, 0, in2_b, xbuf + 2048, 0, 3072, ACT_DIM);
        rms_silu<<<16, 256, 0, stream>>>(xbuf + 0,    3072, in0_g, 1024);
        rms_silu<<<16, 256, 0, stream>>>(xbuf + 1024, 3072, in1_g, 1024);
        rms_silu<<<16, 256, 0, stream>>>(xbuf + 2048, 3072, in2_g, 1024);
        build_xcat<<<2048, 256, 0, stream>>>(xcat, deter, xbuf);
        pack_a<<<dim3(128, 8), 32, 0, stream>>>(xcat, (long long)BB * 4096, 4096, xcatA);
        // blocked GEMMs
        wmma_gemm<<<dim3(64, 8), 32, 0, stream>>>(xcatA, (long long)BB * 4096,
                                                  hid0p, 4096LL * 1024, hid0_b, hbuf, 1024, DETER, 4096);
        rms_silu<<<16, 256, 0, stream>>>(hbuf, DETER, hid0_g, DETER);
        pack_a<<<dim3(32, 8), 32, 0, stream>>>(hbuf, 1024, DETER, hA);
        wmma_gemm<<<dim3(64, 8), 32, 0, stream>>>(hA, (long long)BB * 1024,
                                                  hid1p, 1024LL * 1024, hid1_b, h2buf, 1024, DETER, 1024);
        rms_silu<<<16, 256, 0, stream>>>(h2buf, DETER, hid1_g, DETER);
        pack_a<<<dim3(32, 8), 32, 0, stream>>>(h2buf, 1024, DETER, h2A);
        wmma_gemm<<<dim3(192, 8), 32, 0, stream>>>(h2A, (long long)BB * 1024,
                                                   grup, 1024LL * 3072, gru_b, gbuf, 3072, 24576, 1024);
        gru_update<<<512, 256, 0, stream>>>(gbuf, deter, catbuf, feat, t);
        copy_embed<<<64, 256, 0, stream>>>(catbuf, embed, t);
        // observation head
        pack_a<<<dim3(288, 1), 32, 0, stream>>>(catbuf, 0, 9216, catA);
        wmma_gemm<<<dim3(64, 1), 32, 0, stream>>>(catA, 0, obs0p, 0, obs0_b, hob, 0, 1024, 9216);
        rms_silu<<<16, 256, 0, stream>>>(hob, 1024, obs0_g, 1024);
        pack_a<<<dim3(32, 1), 32, 0, stream>>>(hob, 0, 1024, hobA);
        wmma_gemm<<<dim3(128, 1), 32, 0, stream>>>(hobA, 0, obsLp, 0, obsL_b, lbuf, 0, FLATD, 1024);
        sample_k<<<dim3(BB, 32), 64, 0, stream>>>(lbuf, stoch, feat, t);
    }
}
